// Block_33105607917680
// MI455X (gfx1250) — compile-verified, matrix operand order checked by reading, not run on hardware
//
#include <hip/hip_runtime.h>

// ---------------------------------------------------------------------------
// MI455X (gfx1250) point-transformer block.
//   - GEMMs: bf16 WMMA (v_wmma_f32_16x16x32_bf16), f32 accumulate
//   - B-operand staged into LDS by the Tensor Data Mover (tensor_load_to_lds),
//     double-buffered, synchronized with s_wait_tensorcnt + barriers
//   - bf16 interchange buffers (L2-resident working set)
//   - BN via per-block partials + f32 global atomics, fused apply+residual
// ---------------------------------------------------------------------------

typedef __attribute__((ext_vector_type(16))) __bf16 v16bf;
typedef __attribute__((ext_vector_type(8)))  __bf16 v8bf;
typedef __attribute__((ext_vector_type(8)))  float  v8f;
typedef __attribute__((ext_vector_type(4))) unsigned int u32x4;
typedef __attribute__((ext_vector_type(4))) int          i32x4;
typedef __attribute__((ext_vector_type(8))) int          i32x8;

union V16 { v16bf v; v8bf h[2]; };

__device__ __forceinline__ unsigned short f2bf(float f) {
  unsigned u = __float_as_uint(f);
  unsigned r = u + 0x7FFFu + ((u >> 16) & 1u);   // round-to-nearest-even
  return (unsigned short)(r >> 16);
}
__device__ __forceinline__ float bf2f(unsigned u16) {
  return __uint_as_float(u16 << 16);
}
__device__ __forceinline__ float gelu_tanh(float v) {
  float x3 = v * v * v;
  return 0.5f * v * (1.0f + tanhf(0.7978845608028654f * (v + 0.044715f * x3)));
}

constexpr int BB   = 2;
constexpr int NN   = 8192;
constexpr int KNNK = 16;
constexpr int DIMC = 256;
constexpr int HIDC = 1024;
constexpr int D4C  = 64;
constexpr int M    = BB * NN;            // 16384 rows
constexpr float EPSV = 1e-5f;

// ---------------------------------------------------------------------------
// TDM: DMA one B panel (64 rows x 128 bf16 elements, row stride Kc elements)
// from global memory into LDS at byte offset lds_off. D# per CDNA5 ISA 8.3/8.4:
//   group0: count=1 | lds_addr | global_addr(57b) | type=2
//   group1: data_size=1(2B), tensor_dim0=128, tensor_dim1=64,
//           tile_dim0=128, tile_dim1=64, tensor_dim0_stride=Kc
// ---------------------------------------------------------------------------
__device__ __forceinline__ void tdm_load_b_panel(const unsigned short* gsrc,
                                                 unsigned lds_off, int Kc) {
  unsigned long long ga = (unsigned long long)(uintptr_t)gsrc;
  u32x4 g0;
  g0[0] = 1u;                                            // count=1, user mode
  g0[1] = lds_off;                                       // lds_addr (bytes)
  g0[2] = (unsigned)(ga & 0xffffffffu);                  // global_addr lo
  g0[3] = (unsigned)((ga >> 32) & 0x1ffffffu) | (2u << 30); // addr hi | type=2
  i32x8 g1;
  g1[0] = (int)(1u << 16);                               // data_size = 2 bytes
  g1[1] = (int)(128u << 16);                             // tensor_dim0 lo16
  g1[2] = (int)(64u << 16);                              // dim0 hi | tensor_dim1 lo
  g1[3] = (int)(128u << 16);                             // dim1 hi | tile_dim0=128
  g1[4] = (int)(64u);                                    // tile_dim1=64, tile_dim2=0
  g1[5] = (int)(unsigned)Kc;                             // tensor_dim0_stride lo32
  g1[6] = 0;                                             // stride hi | dim1_stride lo
  g1[7] = 0;
  i32x4 z4 = {0, 0, 0, 0};
#if __has_include(<hip/amd_detail/amd_gfx1250_TDM.h>)
  i32x8 z8 = {0, 0, 0, 0, 0, 0, 0, 0};
  __builtin_amdgcn_tensor_load_to_lds(g0, g1, z4, z4, z8, 0);   // 6-arg toolchain
#else
  __builtin_amdgcn_tensor_load_to_lds(g0, g1, z4, z4, 0);       // ROCm 7.2 (5-arg)
#endif
}

// ---------------------------------------------------------------------------
// fp32 -> bf16 weight convert WITH transpose: W[Kc x Nc] -> Wt[Nc x Kc].
// ---------------------------------------------------------------------------
__global__ void cvt_w_transpose(const float* __restrict__ W,
                                unsigned short* __restrict__ Wt,
                                int Kc, int Nc) {
  int idx = blockIdx.x * blockDim.x + threadIdx.x;
  if (idx >= Kc * Nc) return;
  int k = idx / Nc, n = idx - k * Nc;
  Wt[(size_t)n * Kc + k] = f2bf(W[idx]);
}

// ---------------------------------------------------------------------------
// x init: fp32 copy into running state + bf16 copy for first GEMM.
// ---------------------------------------------------------------------------
__global__ void init_x(const float* __restrict__ xin,
                       float* __restrict__ xs,
                       unsigned short* __restrict__ xbf) {
  int tid = blockIdx.x * blockDim.x + threadIdx.x;
  if (tid >= M * (DIMC / 4)) return;
  float4 v = ((const float4*)xin)[tid];
  ((float4*)xs)[tid] = v;
  uint2 p;
  p.x = (unsigned)f2bf(v.x) | ((unsigned)f2bf(v.y) << 16);
  p.y = (unsigned)f2bf(v.z) | ((unsigned)f2bf(v.w) << 16);
  ((uint2*)xbf)[tid] = p;
}

// ---------------------------------------------------------------------------
// bf16 WMMA GEMM:  out[M x Nc] = A[M x Kc] * Bt[Nc x Kc]^T
// Block = 8 waves = 256 rows x 64 cols. Wave computes 32x64 (8 accumulators).
// B panels (64 x 128 bf16) DMA'd to LDS by TDM, double-buffered; A streamed
// from global (each row touched by exactly one wave).
// mode 0: store fp32   mode 1: +bias, GELU, store bf16   mode 2: store bf16
// ---------------------------------------------------------------------------
__global__ __launch_bounds__(256) void gemm_bf16_wmma(
    const unsigned short* __restrict__ A,
    const unsigned short* __restrict__ Bt,
    float* __restrict__ outF,
    unsigned short* __restrict__ outH,
    const float* __restrict__ bias,
    int Nc, int Kc, int mode) {
  __shared__ __align__(16) unsigned short Bs[2][64 * 128];   // 32 KB, dbl-buffered

  const int lane = threadIdx.x & 31;
  const int wave = threadIdx.x >> 5;
  const int m0 = blockIdx.y * 256 + wave * 32;
  const int n0 = blockIdx.x * 64;
  const int l15  = lane & 15;
  const int half = lane >> 4;
  const int kb   = half * 8;          // K-half selected by lane group (ISA layout)

  const unsigned short* Ap0 = A + (size_t)(m0 + l15) * Kc + kb;
  const unsigned short* Ap1 = A + (size_t)(m0 + 16 + l15) * Kc + kb;
  const unsigned short* Bg  = Bt + (size_t)n0 * Kc;

  v8f acc[2][4] = {};

  const int npanel = Kc >> 7;         // Kc / 128
  int buf = 0;
  if (threadIdx.x == 0) {
    tdm_load_b_panel(Bg, (unsigned)(uintptr_t)&Bs[0][0], Kc);
    __builtin_amdgcn_s_wait_tensorcnt(0);
  }
  __syncthreads();

  for (int p = 0; p < npanel; ++p) {
    if (p + 1 < npanel && threadIdx.x == 0)
      tdm_load_b_panel(Bg + (size_t)(p + 1) * 128,
                       (unsigned)(uintptr_t)&Bs[buf ^ 1][0], Kc);

    const int kbase = p << 7;
    const unsigned short* Bl = &Bs[buf][0];
#pragma unroll
    for (int kk = 0; kk < 128; kk += 32) {
      __builtin_prefetch(Ap0 + kbase + kk + 256, 0, 3);   // near-cache prefetch
      V16 a0, a1;
      a0.h[0] = *(const v8bf*)(Ap0 + kbase + kk);
      a0.h[1] = *(const v8bf*)(Ap0 + kbase + kk + 16);
      a1.h[0] = *(const v8bf*)(Ap1 + kbase + kk);
      a1.h[1] = *(const v8bf*)(Ap1 + kbase + kk + 16);
#pragma unroll
      for (int t = 0; t < 4; ++t) {
        V16 b;
        const unsigned short* bp = Bl + (16 * t + l15) * 128 + kk + kb;
        b.h[0] = *(const v8bf*)(bp);          // ds_load_b128
        b.h[1] = *(const v8bf*)(bp + 16);
        acc[0][t] = __builtin_amdgcn_wmma_f32_16x16x32_bf16(
            false, a0.v, false, b.v, (short)0, acc[0][t], false, false);
        acc[1][t] = __builtin_amdgcn_wmma_f32_16x16x32_bf16(
            false, a1.v, false, b.v, (short)0, acc[1][t], false, false);
      }
    }
    __syncthreads();                   // everyone done reading Bs[buf]
    if (p + 1 < npanel) {
      if (threadIdx.x == 0) __builtin_amdgcn_s_wait_tensorcnt(0);
      __syncthreads();                 // Bs[buf^1] now valid for all waves
      buf ^= 1;
    }
  }

  // C/D layout: VGPR r, lanes 0-15 -> row base+r, lanes 16-31 -> row base+8+r.
#pragma unroll
  for (int s = 0; s < 2; ++s) {
    const int rbase = m0 + s * 16 + half * 8;
#pragma unroll
    for (int t = 0; t < 4; ++t) {
      const int col = n0 + 16 * t + l15;
      const float bv = (mode == 1) ? bias[col] : 0.0f;
#pragma unroll
      for (int r = 0; r < 8; ++r) {
        const int row = rbase + r;
        float v = acc[s][t][r];
        if (mode == 0) {
          outF[(size_t)row * Nc + col] = v;
        } else if (mode == 1) {
          v = gelu_tanh(v + bv);
          outH[(size_t)row * Nc + col] = f2bf(v);
        } else {
          outH[(size_t)row * Nc + col] = f2bf(v);
        }
      }
    }
  }
}

// ---------------------------------------------------------------------------
// LFP gather + Gaussian-kernel aggregation.
// One wave per point; lane owns 8 channels = 2 kernel groups.
// ---------------------------------------------------------------------------
__global__ __launch_bounds__(256) void lfp_gather_agg(
    const unsigned short* __restrict__ hbf,   // [M x 256] bf16 projected feats
    const float* __restrict__ xyz,            // [M x 3]
    const int*   __restrict__ knn,            // [M x 16]
    const float* __restrict__ coor,           // [64 x 3]
    const float* __restrict__ scale,          // [64]
    float* __restrict__ out) {                // [M x 256]
  const int lane = threadIdx.x & 31;
  const int wave = threadIdx.x >> 5;
  const int p = blockIdx.x * 8 + wave;
  const int b = p >> 13;                      // / NN
  const float cx = xyz[p * 3 + 0], cy = xyz[p * 3 + 1], cz = xyz[p * 3 + 2];

  const int g0 = lane * 2, g1 = g0 + 1;
  const float c0x = coor[g0 * 3], c0y = coor[g0 * 3 + 1], c0z = coor[g0 * 3 + 2];
  const float c1x = coor[g1 * 3], c1y = coor[g1 * 3 + 1], c1z = coor[g1 * 3 + 2];
  float s0 = scale[g0]; s0 *= s0;
  float s1 = scale[g1]; s1 *= s1;
  const float cc0 = c0x * c0x + c0y * c0y + c0z * c0z;
  const float cc1 = c1x * c1x + c1y * c1y + c1z * c1z;

  float acc[8] = {0.f, 0.f, 0.f, 0.f, 0.f, 0.f, 0.f, 0.f};
  const int* kp = knn + (size_t)p * KNNK;

#pragma unroll 4
  for (int k = 0; k < KNNK; ++k) {
    const int nb = (b << 13) + kp[k];
    const float px = xyz[nb * 3 + 0] - cx;
    const float py = xyz[nb * 3 + 1] - cy;
    const float pz = xyz[nb * 3 + 2] - cz;
    const float pp = px * px + py * py + pz * pz;
    const float d0 = pp - 2.f * (px * c0x + py * c0y + pz * c0z) + cc0;
    const float d1 = pp - 2.f * (px * c1x + py * c1y + pz * c1z) + cc1;
    const float w0 = __expf(-s0 * d0);
    const float w1 = __expf(-s1 * d1);
    const uint4 u = *(const uint4*)(hbf + (size_t)nb * DIMC + lane * 8);
    acc[0] += bf2f(u.x & 0xffffu) * w0;
    acc[1] += bf2f(u.x >> 16)     * w0;
    acc[2] += bf2f(u.y & 0xffffu) * w0;
    acc[3] += bf2f(u.y >> 16)     * w0;
    acc[4] += bf2f(u.z & 0xffffu) * w1;
    acc[5] += bf2f(u.z >> 16)     * w1;
    acc[6] += bf2f(u.w & 0xffffu) * w1;
    acc[7] += bf2f(u.w >> 16)     * w1;
  }

  float* op = out + (size_t)p * DIMC + lane * 8;
  const float inv = 1.0f / (float)KNNK;
  float4 o0 = {acc[0] * inv, acc[1] * inv, acc[2] * inv, acc[3] * inv};
  float4 o1 = {acc[4] * inv, acc[5] * inv, acc[6] * inv, acc[7] * inv};
  ((float4*)op)[0] = o0;
  ((float4*)op)[1] = o1;
}

// ---------------------------------------------------------------------------
// BatchNorm: zero accumulators, per-block partial sum/sumsq + f32 atomics,
// then fused normalize + gamma/beta + residual add + bf16 recast.
// ---------------------------------------------------------------------------
__global__ void bn_init(float* __restrict__ bnbuf) {
  int t = threadIdx.x;
  if (t < 2 * DIMC) bnbuf[t] = 0.0f;
}

__global__ __launch_bounds__(256) void bn_reduce(const float* __restrict__ h,
                                                 float* __restrict__ bnbuf) {
  const int c = threadIdx.x;                 // 256 channels
  const int r0 = blockIdx.x * 64;
  float s = 0.f, ss = 0.f;
#pragma unroll 8
  for (int r = 0; r < 64; ++r) {
    const float v = h[(size_t)(r0 + r) * DIMC + c];
    s += v; ss += v * v;
  }
  atomicAdd(&bnbuf[c], s);
  atomicAdd(&bnbuf[DIMC + c], ss);
}

__global__ __launch_bounds__(256) void bn_apply_add(
    const float* __restrict__ h, const float* __restrict__ bnbuf,
    const float* __restrict__ gamma, const float* __restrict__ beta,
    float* __restrict__ x, unsigned short* __restrict__ xbf) {
  const int tid = blockIdx.x * blockDim.x + threadIdx.x;   // per 4 channels
  if (tid >= M * (DIMC / 4)) return;
  const int c = (tid & (DIMC / 4 - 1)) * 4;
  const float4 hv = ((const float4*)h)[tid];
  const float4 xv = ((const float4*)x)[tid];
  const float inv = 1.0f / (float)M;
  float4 o;
#pragma unroll
  for (int j = 0; j < 4; ++j) {
    const float hj = (&hv.x)[j];
    const float mu = bnbuf[c + j] * inv;
    const float var = bnbuf[DIMC + c + j] * inv - mu * mu;
    const float rs = rsqrtf(var + EPSV);
    (&o.x)[j] = (&xv.x)[j] + (hj - mu) * rs * gamma[c + j] + beta[c + j];
  }
  ((float4*)x)[tid] = o;
  uint2 p;
  p.x = (unsigned)f2bf(o.x) | ((unsigned)f2bf(o.y) << 16);
  p.y = (unsigned)f2bf(o.z) | ((unsigned)f2bf(o.w) << 16);
  ((uint2*)xbf)[tid] = p;
}

// ---------------------------------------------------------------------------
// Host orchestration
// ---------------------------------------------------------------------------
extern "C" void kernel_launch(void* const* d_in, const int* in_sizes, int n_in,
                              void* d_out, int out_size, void* d_ws, size_t ws_size,
                              hipStream_t stream) {
  const float* x_in      = (const float*)d_in[0];
  const float* xyz       = (const float*)d_in[1];
  const int*   knn       = (const int*)d_in[2];
  const float* lfp_proj  = (const float*)d_in[3];
  const float* lfp_coor  = (const float*)d_in[4];
  const float* lfp_scale = (const float*)d_in[5];
  const float* lfp_gamma = (const float*)d_in[6];
  const float* lfp_beta  = (const float*)d_in[7];
  const float* mlp_w1    = (const float*)d_in[8];
  const float* mlp_b1    = (const float*)d_in[9];
  const float* mlp_w2    = (const float*)d_in[10];
  const float* mlp_gamma = (const float*)d_in[11];
  const float* mlp_beta  = (const float*)d_in[12];

  float* xs = (float*)d_out;                           // running state [M x 256]

  char* ws = (char*)d_ws;
  size_t off = 0;
  auto alloc = [&](size_t bytes) {
    char* p = ws + off;
    off = (off + bytes + 255) & ~(size_t)255;
    return p;
  };
  unsigned short* xbf  = (unsigned short*)alloc((size_t)M * DIMC * 2);   //  8 MB
  unsigned short* hbf  = (unsigned short*)alloc((size_t)M * HIDC * 2);   // 32 MB
  float*          tmp  = (float*)alloc((size_t)M * DIMC * 4);            // 16 MB
  unsigned short* w1t  = (unsigned short*)alloc((size_t)HIDC * DIMC * 2);
  unsigned short* w2t  = (unsigned short*)alloc((size_t)DIMC * HIDC * 2);
  unsigned short* pjt  = (unsigned short*)alloc((size_t)DIMC * DIMC * 2);
  float*          bnb  = (float*)alloc(2 * DIMC * 4);

  const int elem4 = M * (DIMC / 4);

  auto gemm = [&](const unsigned short* A, const unsigned short* Bt,
                  float* oF, unsigned short* oH, const float* bias,
                  int Nc, int Kc, int mode) {
    dim3 grid(Nc / 64, M / 256);
    gemm_bf16_wmma<<<grid, 256, 0, stream>>>(A, Bt, oF, oH, bias, Nc, Kc, mode);
  };
  auto bn = [&](const float* gamma, const float* beta) {
    bn_init<<<1, 2 * DIMC, 0, stream>>>(bnb);
    bn_reduce<<<M / 64, DIMC, 0, stream>>>(tmp, bnb);
    bn_apply_add<<<(elem4 + 255) / 256, 256, 0, stream>>>(tmp, bnb, gamma, beta, xs, xbf);
  };
  auto mlp = [&](int j) {
    cvt_w_transpose<<<(DIMC * HIDC + 255) / 256, 256, 0, stream>>>(
        mlp_w1 + (size_t)j * DIMC * HIDC, w1t, DIMC, HIDC);
    gemm(xbf, w1t, nullptr, hbf, mlp_b1 + (size_t)j * HIDC, HIDC, DIMC, /*bias+gelu+bf16*/1);
    cvt_w_transpose<<<(HIDC * DIMC + 255) / 256, 256, 0, stream>>>(
        mlp_w2 + (size_t)j * HIDC * DIMC, w2t, HIDC, DIMC);
    gemm(hbf, w2t, tmp, nullptr, nullptr, DIMC, HIDC, /*fp32*/0);
    bn(mlp_gamma + (size_t)j * DIMC, mlp_beta + (size_t)j * DIMC);
  };
  auto lfp = [&](int i) {
    cvt_w_transpose<<<(DIMC * DIMC + 255) / 256, 256, 0, stream>>>(
        lfp_proj + (size_t)i * DIMC * DIMC, pjt, DIMC, DIMC);
    gemm(xbf, pjt, nullptr, hbf, nullptr, DIMC, DIMC, /*bf16*/2);
    lfp_gather_agg<<<M / 8, 256, 0, stream>>>(
        hbf, xyz, knn, lfp_coor + (size_t)i * D4C * 3, lfp_scale + (size_t)i * D4C, tmp);
    bn(lfp_gamma + (size_t)i * DIMC, lfp_beta + (size_t)i * DIMC);
  };

  init_x<<<(elem4 + 255) / 256, 256, 0, stream>>>(x_in, xs, xbf);
  mlp(0);
  lfp(0);
  lfp(1);
  mlp(1);
  lfp(2);
  lfp(3);
  mlp(2);
}
